// CustomMultiheadAttention_GRAM_77592879170093
// MI455X (gfx1250) — matrix operand order repaired and checked by reading, not compiled, not run on hardware
//
#include <hip/hip_runtime.h>
#include <hip/hip_bf16.h>
#include <cstdint>
#include <cstddef>

// ---------------- problem constants ----------------
#define E_DIM  1024
#define H_NUM  16
#define DH     64
#define L_DIM  512
#define B_DIM  4
#define S_DIM  1024
#define BH_NUM (B_DIM * H_NUM)
#define HPB    4                  // heads per attention block
#define HG_NUM (H_NUM / HPB)      // head groups
#define GRAM_W 1.5f
#define EPS_G  1e-8f

// ---------------- WMMA types ----------------
typedef __attribute__((ext_vector_type(16))) __bf16 v16bf;
typedef __attribute__((ext_vector_type(8)))  float  v8f;

union Frag {
  v16bf bf;
  unsigned short u[16];
  uint4 q[2];
};

__device__ __forceinline__ unsigned short f2bf(float f) {
  unsigned u = __float_as_uint(f);
  u += 0x7FFFu + ((u >> 16) & 1u);      // round-to-nearest-even
  return (unsigned short)(u >> 16);
}
__device__ __forceinline__ float bf2f(unsigned short h) {
  return __uint_as_float(((unsigned)h) << 16);
}

// ---------------- CDNA5 async global->LDS (ASYNCcnt path) ----------------
// Builtin signature (from hipcc diagnostic): param0 = v4i* in global AS,
// param1 = v4i* in LDS AS, then imm offset, imm cpol.
typedef int v4i_ __attribute__((vector_size(16)));
typedef __attribute__((address_space(1))) v4i_ g_v4i;   // global 128-bit
typedef __attribute__((address_space(3))) v4i_ l_v4i;   // LDS 128-bit

#if defined(__has_builtin)
#  if __has_builtin(__builtin_amdgcn_global_load_async_to_lds_b128) && \
      __has_builtin(__builtin_amdgcn_s_wait_asynccnt)
#    define HAVE_ASYNC_LDS 1
#  endif
#endif
#ifndef HAVE_ASYNC_LDS
#  define HAVE_ASYNC_LDS 0
#endif

__device__ __forceinline__ void lds_copy16(unsigned short* dst_lds,
                                           const unsigned short* src_glb) {
#if HAVE_ASYNC_LDS
  __builtin_amdgcn_global_load_async_to_lds_b128(
      (g_v4i*)src_glb, (l_v4i*)dst_lds, 0, 0);
#else
  *(uint4*)dst_lds = *(const uint4*)src_glb;
#endif
}
__device__ __forceinline__ void async_wait_all() {
#if HAVE_ASYNC_LDS
  __builtin_amdgcn_s_wait_asynccnt(0);
#endif
}

// =====================================================================
// GEMM: out[M,N] = (A[M,K] @ W[K,N] + bias[N]) * scale
// A,W fp32 in HBM; converted to bf16 in LDS; WMMA f32_16x16x32_bf16.
// OUT_MODE 0 -> bf16 out, OUT_MODE 1 -> f32 out.
// Requires M%128==0, N%64==0, K%32==0 (true for all uses here).
// =====================================================================
#define BM 128
#define BN 64
#define BK 32
#define LPAD 8   // keep rows 16B-aligned (stride 40 ushorts = 80B)

template <int OUT_MODE>
__global__ __launch_bounds__(256) void gemm_bias_kernel(
    const float* __restrict__ A, const float* __restrict__ W,
    const float* __restrict__ bias, void* __restrict__ out,
    int M, int N, int K, float scale)
{
  __shared__ __align__(16) unsigned short As [BM][BK + LPAD];
  __shared__ __align__(16) unsigned short WsT[BN][BK + LPAD];

  const int tid  = threadIdx.x;
  const int wid  = tid >> 5;
  const int lane = tid & 31;
  const int half = lane >> 4;
  const int l16  = lane & 15;
  const int m0   = blockIdx.y * BM;
  const int n0   = blockIdx.x * BN;
  const int wm   = wid & 3;   // 4 wave-rows of 32
  const int wn   = wid >> 2;  // 2 wave-cols of 32

  v8f acc[2][2];
  #pragma unroll
  for (int i = 0; i < 2; ++i)
    #pragma unroll
    for (int j = 0; j < 2; ++j) acc[i][j] = v8f{};

  for (int k0 = 0; k0 < K; k0 += BK) {
    // prefetch next K tile into the cache hierarchy while this one computes
    if (k0 + BK < K) {
      __builtin_prefetch(A + (size_t)(m0 + (tid >> 1)) * K + (k0 + BK), 0, 3);
      __builtin_prefetch(W + (size_t)(k0 + BK + (tid >> 4)) * N + n0, 0, 3);
    }
    // A tile: 128x32 f32 -> bf16 LDS (1024 float4 chunks)
    #pragma unroll
    for (int i = 0; i < 4; ++i) {
      int c = tid + i * 256;
      int row = c >> 3, col = (c & 7) << 2;
      const float4 v = *(const float4*)(A + (size_t)(m0 + row) * K + k0 + col);
      As[row][col + 0] = f2bf(v.x); As[row][col + 1] = f2bf(v.y);
      As[row][col + 2] = f2bf(v.z); As[row][col + 3] = f2bf(v.w);
    }
    // W tile: 32x64 f32 -> transposed bf16 LDS (512 float4 chunks)
    #pragma unroll
    for (int i = 0; i < 2; ++i) {
      int c = tid + i * 256;
      int kr = c >> 4, nc = (c & 15) << 2;
      const float4 v = *(const float4*)(W + (size_t)(k0 + kr) * N + n0 + nc);
      WsT[nc + 0][kr] = f2bf(v.x); WsT[nc + 1][kr] = f2bf(v.y);
      WsT[nc + 2][kr] = f2bf(v.z); WsT[nc + 3][kr] = f2bf(v.w);
    }
    __syncthreads();

    Frag a[2], bf[2];
    #pragma unroll
    for (int mt = 0; mt < 2; ++mt) {
      int row = wm * 32 + mt * 16 + l16;
      // A 16x32 bf16 layout: elem j -> K = half*16 + j
      a[mt].q[0] = *(const uint4*)(&As[row][half * 16]);
      a[mt].q[1] = *(const uint4*)(&As[row][half * 16 + 8]);
    }
    #pragma unroll
    for (int nt = 0; nt < 2; ++nt) {
      int col = wn * 32 + nt * 16 + l16;
      // B 32x16 layout: elem j -> K = half*16 + j, N = lane&15
      bf[nt].q[0] = *(const uint4*)(&WsT[col][half * 16]);
      bf[nt].q[1] = *(const uint4*)(&WsT[col][half * 16 + 8]);
    }
    #pragma unroll
    for (int mt = 0; mt < 2; ++mt)
      #pragma unroll
      for (int nt = 0; nt < 2; ++nt)
        acc[mt][nt] = __builtin_amdgcn_wmma_f32_16x16x32_bf16(
            false, a[mt].bf, false, bf[nt].bf, (short)0, acc[mt][nt], false, false);
    __syncthreads();
  }

  #pragma unroll
  for (int mt = 0; mt < 2; ++mt) {
    #pragma unroll
    for (int nt = 0; nt < 2; ++nt) {
      int n = n0 + wn * 32 + nt * 16 + l16;
      float bv = bias[n];
      #pragma unroll
      for (int r = 0; r < 8; ++r) {
        int m = m0 + wm * 32 + mt * 16 + r + half * 8;
        float v = (acc[mt][nt][r] + bv) * scale;
        if (OUT_MODE == 0)
          ((unsigned short*)out)[(size_t)m * N + n] = f2bf(v);
        else
          ((float*)out)[(size_t)m * N + n] = v;
      }
    }
  }
}

// =====================================================================
// Gram statistics
// =====================================================================
__global__ __launch_bounds__(256) void kstats_kernel(
    const unsigned short* __restrict__ k1, const unsigned short* __restrict__ k2,
    float* __restrict__ vv, float* __restrict__ aa, float* __restrict__ va)
{
  int idx = blockIdx.x * 256 + threadIdx.x;
  if (idx >= BH_NUM * S_DIM) return;
  int bh = idx >> 10, s = idx & (S_DIM - 1);
  int b = bh >> 4, h = bh & 15;
  const unsigned short* p1 = k1 + ((size_t)(s * B_DIM + b)) * E_DIM + h * DH;
  const unsigned short* p2 = k2 + ((size_t)(s * B_DIM + b)) * E_DIM + h * DH;
  float svv = 0.f, saa = 0.f, sva = 0.f;
  #pragma unroll 8
  for (int d = 0; d < DH; ++d) {
    float f1 = bf2f(p1[d]), f2 = bf2f(p2[d]);
    svv += f1 * f1; saa += f2 * f2; sva += f1 * f2;
  }
  vv[idx] = svv; aa[idx] = saa; va[idx] = sva;
}

__global__ __launch_bounds__(256) void ll_kernel(
    const unsigned short* __restrict__ q, float* __restrict__ ll)
{
  int idx = blockIdx.x * 256 + threadIdx.x;
  if (idx >= BH_NUM * L_DIM) return;
  int bh = idx >> 9, l = idx & (L_DIM - 1);
  int b = bh >> 4, h = bh & 15;
  const unsigned short* p = q + ((size_t)(l * B_DIM + b)) * E_DIM + h * DH;
  float s = 0.f;
  #pragma unroll 8
  for (int d = 0; d < DH; ++d) { float f = bf2f(p[d]); s += f * f; }
  ll[idx] = s;
}

// =====================================================================
// Attention: block = (l-tile of 32, b, head-group of 4); loops 4 heads.
//  Phase A: logits = GRAM_W*gram + q@k1^T + q@k2^T via WMMA + epilogue
//  softmax over full S in LDS; per-head-group avg partial (single owner)
//  Phase B: out1 = attn@v1, out2 = attn@v2 via WMMA
// =====================================================================
#define BLQ 32
#define SCH 128

__global__ __launch_bounds__(256) void attention_kernel(
    const unsigned short* __restrict__ qact,
    const unsigned short* __restrict__ k1a, const unsigned short* __restrict__ k2a,
    const unsigned short* __restrict__ v1a, const unsigned short* __restrict__ v2a,
    const float* __restrict__ llg,
    const float* __restrict__ vvg, const float* __restrict__ aag,
    const float* __restrict__ vag,
    float* __restrict__ out1w, float* __restrict__ out2w,
    float* __restrict__ avg_part)
{
  __shared__ __align__(16) float          probs[BLQ][S_DIM];     // 128 KB
  __shared__ __align__(16) unsigned short stage[2][SCH * DH];    // 32 KB
  __shared__ __align__(16) unsigned short qlds [BLQ * DH];       // 4 KB
  __shared__ float lls[BLQ];

  const int tid  = threadIdx.x;
  const int wid  = tid >> 5;
  const int lane = tid & 31;
  const int half = lane >> 4;
  const int l16  = lane & 15;
  const int l0   = blockIdx.x * BLQ;
  const int b    = blockIdx.y;
  const int hg   = blockIdx.z;

  // this head-group's private partial plane of the head-mean
  float* avgp = avg_part + (size_t)hg * ((size_t)B_DIM * L_DIM * S_DIM);

  for (int hl = 0; hl < HPB; ++hl) {
    const int h  = hg * HPB + hl;
    const int bh = b * H_NUM + h;
    // load q tile [32][64] bf16 (each thread: one 16B async chunk)
    {
      int f  = tid * 8;
      int lq = f >> 6, d = f & 63;
      lds_copy16(&qlds[f],
                 qact + ((size_t)((l0 + lq) * B_DIM + b)) * E_DIM + h * DH + d);
    }
    if (tid < BLQ) lls[tid] = llg[(size_t)bh * L_DIM + l0 + tid];
    async_wait_all();
    __syncthreads();

    // ---------------- Phase A: logits ----------------
    for (int sc0 = 0; sc0 < S_DIM; sc0 += SCH) {
      #pragma unroll
      for (int i = 0; i < 8; ++i) {          // stage k1,k2 chunk [128][64]
        int c = tid + i * 256;
        int which = c >> 10, cc = c & 1023;
        int srow = cc >> 3, dpart = (cc & 7) << 3;
        const unsigned short* src = which ? k2a : k1a;
        lds_copy16(&stage[which][srow * DH + dpart],
                   src + ((size_t)((sc0 + srow) * B_DIM + b)) * E_DIM + h * DH + dpart);
      }
      async_wait_all();
      __syncthreads();

      v8f s1[2], s2[2];
      #pragma unroll
      for (int mt = 0; mt < 2; ++mt) { s1[mt] = v8f{}; s2[mt] = v8f{}; }

      #pragma unroll
      for (int kk = 0; kk < 2; ++kk) {       // K = 64 = 2 x 32
        Frag a[2], b1, b2;
        #pragma unroll
        for (int mt = 0; mt < 2; ++mt) {
          int row = mt * 16 + l16;
          a[mt].q[0] = *(const uint4*)(&qlds[row * DH + kk * 32 + half * 16]);
          a[mt].q[1] = *(const uint4*)(&qlds[row * DH + kk * 32 + half * 16 + 8]);
        }
        int srow = wid * 16 + l16;           // wave = one 16-wide s tile
        b1.q[0] = *(const uint4*)(&stage[0][srow * DH + kk * 32 + half * 16]);
        b1.q[1] = *(const uint4*)(&stage[0][srow * DH + kk * 32 + half * 16 + 8]);
        b2.q[0] = *(const uint4*)(&stage[1][srow * DH + kk * 32 + half * 16]);
        b2.q[1] = *(const uint4*)(&stage[1][srow * DH + kk * 32 + half * 16 + 8]);
        #pragma unroll
        for (int mt = 0; mt < 2; ++mt) {
          s1[mt] = __builtin_amdgcn_wmma_f32_16x16x32_bf16(
              false, a[mt].bf, false, b1.bf, (short)0, s1[mt], false, false);
          s2[mt] = __builtin_amdgcn_wmma_f32_16x16x32_bf16(
              false, a[mt].bf, false, b2.bf, (short)0, s2[mt], false, false);
        }
      }
      // gram + logit epilogue directly on accumulators
      {
        int sidx = sc0 + wid * 16 + l16;
        float vv  = vvg[(size_t)bh * S_DIM + sidx];
        float aaV = aag[(size_t)bh * S_DIM + sidx];
        float vaV = vag[(size_t)bh * S_DIM + sidx];
        #pragma unroll
        for (int mt = 0; mt < 2; ++mt) {
          #pragma unroll
          for (int r = 0; r < 8; ++r) {
            int m = mt * 16 + r + half * 8;
            float lv = s1[mt][r], la = s2[mt][r];
            float llv = lls[m];
            float det = llv * (vv * aaV - vaV * vaV)
                      - lv * (lv * aaV - la * vaV)
                      + la * (lv * vaV - la * vv);
            det = fmaxf(det, EPS_G);
            probs[m][sidx] = GRAM_W * (-sqrtf(det)) + lv + la;
          }
        }
      }
      __syncthreads();
    }

    // ---------------- softmax (wave handles 4 rows) ----------------
    #pragma unroll
    for (int rr = 0; rr < 4; ++rr) {
      int row = wid * 4 + rr;
      float mx = -3.4e38f;
      for (int j = lane; j < S_DIM; j += 32) mx = fmaxf(mx, probs[row][j]);
      #pragma unroll
      for (int o = 16; o > 0; o >>= 1) mx = fmaxf(mx, __shfl_xor(mx, o, 32));
      float sum = 0.f;
      for (int j = lane; j < S_DIM; j += 32) {
        float e = __expf(probs[row][j] - mx);
        probs[row][j] = e; sum += e;
      }
      #pragma unroll
      for (int o = 16; o > 0; o >>= 1) sum += __shfl_xor(sum, o, 32);
      float inv = 1.f / sum;
      for (int j = lane; j < S_DIM; j += 32) probs[row][j] *= inv;
    }
    __syncthreads();

    // per-head-group partial of the head mean (sole owner of this plane slice)
    for (int c = tid; c < BLQ * S_DIM; c += 256) {
      int row = c >> 10, s = c & (S_DIM - 1);
      float p = probs[row][s];
      float* dst = avgp + ((size_t)(b * L_DIM + l0 + row)) * S_DIM + s;
      if (hl == 0) *dst = p; else *dst += p;
    }

    // ---------------- Phase B: attn @ v1, attn @ v2 ----------------
    const int wm = wid & 1;   // 2 m-tiles
    const int wn = wid >> 1;  // 4 n-tiles (DH = 64)
    v8f o1 = v8f{}, o2 = v8f{};
    for (int sc0 = 0; sc0 < S_DIM; sc0 += SCH) {
      #pragma unroll
      for (int i = 0; i < 8; ++i) {          // stage v1,v2 transposed: [d][s]
        int c = tid + i * 256;
        int which = c >> 10, cc = c & 1023;
        int srow = cc >> 3, dpart = (cc & 7) << 3;
        const unsigned short* src = which ? v2a : v1a;
        uint4 vq = *(const uint4*)(src + ((size_t)((sc0 + srow) * B_DIM + b)) * E_DIM + h * DH + dpart);
        const unsigned short* us = (const unsigned short*)&vq;
        #pragma unroll
        for (int j = 0; j < 8; ++j)
          stage[which][(dpart + j) * SCH + srow] = us[j];
      }
      __syncthreads();

      #pragma unroll
      for (int kk = 0; kk < 4; ++kk) {       // 128 = 4 x 32 K-steps
        Frag a, bb1, bb2;
        int row = wm * 16 + l16;
        #pragma unroll
        for (int j = 0; j < 16; ++j) {
          int soff = half * 16 + j;
          a.u[j] = f2bf(probs[row][sc0 + kk * 32 + soff]);
        }
        int d = wn * 16 + l16;
        bb1.q[0] = *(const uint4*)(&stage[0][d * SCH + kk * 32 + half * 16]);
        bb1.q[1] = *(const uint4*)(&stage[0][d * SCH + kk * 32 + half * 16 + 8]);
        bb2.q[0] = *(const uint4*)(&stage[1][d * SCH + kk * 32 + half * 16]);
        bb2.q[1] = *(const uint4*)(&stage[1][d * SCH + kk * 32 + half * 16 + 8]);
        o1 = __builtin_amdgcn_wmma_f32_16x16x32_bf16(
            false, a.bf, false, bb1.bf, (short)0, o1, false, false);
        o2 = __builtin_amdgcn_wmma_f32_16x16x32_bf16(
            false, a.bf, false, bb2.bf, (short)0, o2, false, false);
      }
      __syncthreads();
    }
    // write merged-head tiles: [L,B,E] with e = h*64+d
    {
      int dcol = h * DH + wn * 16 + l16;
      #pragma unroll
      for (int r = 0; r < 8; ++r) {
        int lrow = l0 + wm * 16 + r + half * 8;
        size_t off = ((size_t)(lrow * B_DIM + b)) * E_DIM + dcol;
        out1w[off] = o1[r];
        out2w[off] = o2[r];
      }
    }
    __syncthreads();
  }
}

// =====================================================================
// head-mean reduction: avg = (sum of HG_NUM partials) / H
// =====================================================================
__global__ __launch_bounds__(256) void avg_reduce_kernel(
    const float* __restrict__ part, float* __restrict__ out, int n)
{
  int idx = blockIdx.x * 256 + threadIdx.x;
  if (idx >= n) return;
  const size_t plane = (size_t)B_DIM * L_DIM * S_DIM;
  float s = part[idx] + part[plane + idx] + part[2 * plane + idx] + part[3 * plane + idx];
  out[idx] = s * (1.0f / (float)H_NUM);
}

// =====================================================================
// gate combine: z = 0.5*(out1*sigmoid(g1) + out2*sigmoid(g2))
// =====================================================================
__global__ __launch_bounds__(256) void gate_combine_kernel(
    const float* __restrict__ out1, const float* __restrict__ out2,
    const float* __restrict__ g1, const float* __restrict__ g2,
    float* __restrict__ z, int n)
{
  int idx = blockIdx.x * 256 + threadIdx.x;
  if (idx >= n) return;
  float s1 = 1.f / (1.f + __expf(-g1[idx]));
  float s2 = 1.f / (1.f + __expf(-g2[idx]));
  z[idx] = 0.5f * (out1[idx] * s1 + out2[idx] * s2);
}

// =====================================================================
// host-side orchestration
// =====================================================================
extern "C" void kernel_launch(void* const* d_in, const int* in_sizes, int n_in,
                              void* d_out, int out_size, void* d_ws, size_t ws_size,
                              hipStream_t stream) {
  (void)in_sizes; (void)n_in; (void)out_size; (void)ws_size;

  const float* query = (const float*)d_in[0];
  const float* mod1  = (const float*)d_in[1];
  const float* mod2  = (const float*)d_in[2];
  const float* q_w  = (const float*)d_in[3];  const float* q_b  = (const float*)d_in[4];
  const float* k1_w = (const float*)d_in[5];  const float* k1_b = (const float*)d_in[6];
  const float* k2_w = (const float*)d_in[7];  const float* k2_b = (const float*)d_in[8];
  const float* v1_w = (const float*)d_in[9];  const float* v1_b = (const float*)d_in[10];
  const float* v2_w = (const float*)d_in[11]; const float* v2_b = (const float*)d_in[12];
  const float* g1_w = (const float*)d_in[13]; const float* g1_b = (const float*)d_in[14];
  const float* g2_w = (const float*)d_in[15]; const float* g2_b = (const float*)d_in[16];
  const float* o_w  = (const float*)d_in[17]; const float* o_b  = (const float*)d_in[18];

  const int MQ = L_DIM * B_DIM;   // 2048
  const int MS = S_DIM * B_DIM;   // 4096

  // workspace layout (256B aligned)
  char* ws = (char*)d_ws;
  size_t off = 0;
  auto alloc = [&](size_t bytes) {
    size_t o = off; off += (bytes + 255) & ~(size_t)255; return o;
  };
  unsigned short* q_act  = (unsigned short*)(ws + alloc((size_t)MQ * E_DIM * 2));
  unsigned short* k1_act = (unsigned short*)(ws + alloc((size_t)MS * E_DIM * 2));
  unsigned short* k2_act = (unsigned short*)(ws + alloc((size_t)MS * E_DIM * 2));
  unsigned short* v1_act = (unsigned short*)(ws + alloc((size_t)MS * E_DIM * 2));
  unsigned short* v2_act = (unsigned short*)(ws + alloc((size_t)MS * E_DIM * 2));
  float* g1_act = (float*)(ws + alloc((size_t)MQ * E_DIM * 4));
  float* g2_act = (float*)(ws + alloc((size_t)MQ * E_DIM * 4));
  float* out1w  = (float*)(ws + alloc((size_t)MQ * E_DIM * 4));
  float* out2w  = (float*)(ws + alloc((size_t)MQ * E_DIM * 4));
  float* z_ws   = (float*)(ws + alloc((size_t)MQ * E_DIM * 4));
  float* llb    = (float*)(ws + alloc((size_t)BH_NUM * L_DIM * 4));
  float* vvb    = (float*)(ws + alloc((size_t)BH_NUM * S_DIM * 4));
  float* aab    = (float*)(ws + alloc((size_t)BH_NUM * S_DIM * 4));
  float* vab    = (float*)(ws + alloc((size_t)BH_NUM * S_DIM * 4));
  float* avgp   = (float*)(ws + alloc((size_t)HG_NUM * B_DIM * L_DIM * S_DIM * 4));

  float* attn_out = (float*)d_out;                               // [L,B,E]
  float* avg_out  = (float*)d_out + (size_t)MQ * E_DIM;          // [B,L,S]

  const float scaling = 0.125f;  // DH^-0.5

  dim3 blk(256);
  dim3 gq(E_DIM / BN, MQ / BM);
  dim3 gs(E_DIM / BN, MS / BM);

  // 1) projections
  gemm_bias_kernel<0><<<gq, blk, 0, stream>>>(query, q_w,  q_b,  q_act,  MQ, E_DIM, E_DIM, scaling);
  gemm_bias_kernel<0><<<gs, blk, 0, stream>>>(mod1,  k1_w, k1_b, k1_act, MS, E_DIM, E_DIM, 1.0f);
  gemm_bias_kernel<0><<<gs, blk, 0, stream>>>(mod2,  k2_w, k2_b, k2_act, MS, E_DIM, E_DIM, 1.0f);
  gemm_bias_kernel<0><<<gs, blk, 0, stream>>>(mod1,  v1_w, v1_b, v1_act, MS, E_DIM, E_DIM, 1.0f);
  gemm_bias_kernel<0><<<gs, blk, 0, stream>>>(mod2,  v2_w, v2_b, v2_act, MS, E_DIM, E_DIM, 1.0f);
  gemm_bias_kernel<1><<<gq, blk, 0, stream>>>(query, g1_w, g1_b, g1_act, MQ, E_DIM, E_DIM, 1.0f);
  gemm_bias_kernel<1><<<gq, blk, 0, stream>>>(query, g2_w, g2_b, g2_act, MQ, E_DIM, E_DIM, 1.0f);

  // 2) gram statistics
  ll_kernel<<<dim3((BH_NUM * L_DIM) / 256), blk, 0, stream>>>(q_act, llb);
  kstats_kernel<<<dim3((BH_NUM * S_DIM) / 256), blk, 0, stream>>>(k1_act, k2_act, vvb, aab, vab);

  // 3) attention (scores + gram + softmax + avg-partials + attn@v)
  attention_kernel<<<dim3(L_DIM / BLQ, B_DIM, HG_NUM), blk, 0, stream>>>(
      q_act, k1_act, k2_act, v1_act, v2_act,
      llb, vvb, aab, vab, out1w, out2w, avgp);

  // 4) head-mean reduction into d_out
  {
    int n = B_DIM * L_DIM * S_DIM;
    avg_reduce_kernel<<<dim3((n + 255) / 256), blk, 0, stream>>>(avgp, avg_out, n);
  }

  // 5) gating combine
  {
    int n = MQ * E_DIM;
    gate_combine_kernel<<<dim3((n + 255) / 256), blk, 0, stream>>>(
        out1w, out2w, g1_act, g2_act, z_ws, n);
  }

  // 6) output projection -> d_out
  gemm_bias_kernel<1><<<gq, blk, 0, stream>>>(z_ws, o_w, o_b, attn_out, MQ, E_DIM, E_DIM, 1.0f);
}
